// PPI_attn_66597762891844
// MI455X (gfx1250) — matrix operand order repaired and checked by reading, not compiled, not run on hardware
//
#include <hip/hip_runtime.h>
#include <hip/hip_bf16.h>

#define DEVI __device__ __forceinline__

typedef __bf16 bf16_t;
typedef __bf16 v16bf __attribute__((ext_vector_type(16)));
typedef __bf16 v8bf  __attribute__((ext_vector_type(8)));
typedef float  v8f   __attribute__((ext_vector_type(8)));

// ---------------- model dims ----------------
constexpr int L = 512, B = 64, E = 200, EP = 224, H = 256;
constexpr int NC4  = 3328;          // combined projection columns: 1024+1024+1024(pad iou)+256
constexpr int ROWS = L * B;         // 32768

// ---------------- workspace layout (bytes) ----------------
constexpr size_t SZ_EMB  = (size_t)ROWS * EP * 2;
constexpr size_t SZ_WBIG = (size_t)NC4 * EP * 2;
constexpr size_t SZ_BIAS = (size_t)NC4 * 4;
constexpr size_t SZ_C    = (size_t)ROWS * NC4 * 2;
constexpr size_t SZ_WHH  = 2ull * 1024 * 256 * 2;
constexpr size_t SZ_UIOU = 1024ull * 256 * 2;
constexpr size_t SZ_UTF  = 256ull * 256 * 2;
constexpr size_t SZ_OUT  = (size_t)L * B * H * 4;
constexpr size_t SZ_SUM  = (size_t)B * L * H * 4;
constexpr size_t SZ_CST  = 2ull * B * H * 4;

constexpr size_t OFF_EMB  = 0;
constexpr size_t OFF_WBIG = OFF_EMB  + SZ_EMB;
constexpr size_t OFF_BIAS = OFF_WBIG + SZ_WBIG;
constexpr size_t OFF_C    = OFF_BIAS + SZ_BIAS;
constexpr size_t OFF_WHH  = OFF_C    + SZ_C;
constexpr size_t OFF_UIOU = OFF_WHH  + SZ_WHH;
constexpr size_t OFF_UTF  = OFF_UIOU + SZ_UIOU;
constexpr size_t OFF_OUTF = OFF_UTF  + SZ_UTF;
constexpr size_t OFF_OUTB = OFF_OUTF + SZ_OUT;
constexpr size_t OFF_HSUM = OFF_OUTB + SZ_OUT;
constexpr size_t OFF_CSUM = OFF_HSUM + SZ_SUM;   // contiguous with HSUM (zeroed together)
constexpr size_t OFF_CST  = OFF_CSUM + SZ_SUM;
constexpr size_t OFF_TH   = OFF_CST  + SZ_CST;

// ---------------- helpers ----------------
DEVI float sigf(float x)     { return 1.f / (1.f + __expf(-x)); }
DEVI float tanhfast(float x) { float e = __expf(2.f * x); return 1.f - 2.f / (e + 1.f); }

union FragU { v16bf v; v8bf h[2]; };

// A-matrix 16x32 bf16 fragment (ISA 7.12.2): lane<16 -> row=lane, K={kb..kb+7, kb+16..kb+23};
// lane>=16 -> K shifted by 8.  'row' points at the lane's row, hi = lane>>4.
DEVI v16bf loadA(const bf16_t* row, int kb, int hi) {
  FragU f; const bf16_t* p = row + kb * 32 + (hi << 3);
  f.h[0] = *(const v8bf*)p; f.h[1] = *(const v8bf*)(p + 16); return f.v;
}
// B-matrix 32x16 bf16 fragment: lane<16 -> col=lane, K={kb..kb+7, kb+16..kb+23}; lane>=16 -> K shifted by 8.
// 'row' points at column n of the pre-transposed weight [N][K].
DEVI v16bf loadB(const bf16_t* row, int kb, int hi) {
  FragU f; const bf16_t* p = row + kb * 32 + (hi << 3);
  f.h[0] = *(const v8bf*)p; f.h[1] = *(const v8bf*)(p + 16); return f.v;
}
DEVI v8f wmma_bf16(v16bf a, v16bf b, v8f c) {
  return __builtin_amdgcn_wmma_f32_16x16x32_bf16(false, a, false, b, (short)0, c, false, false);
}

// ---------------- weight prep ----------------
// Combined input-projection weight, transposed [N=3328][K=224], gate-interleaved columns.
// n<1024: W_ih_f (n=4j+g, g in i,f,u,o); 1024..2047: W_ih_b; 2048..3071: W_iou padded (i,o,u,0); 3072..: W_tf
__global__ void k_prep_wbig(const float* Wihf, const float* bf_, const float* Wihb, const float* bb_,
                            const float* Wiou, const float* biou, const float* Wtf, const float* btf,
                            bf16_t* Wt, float* bias) {
  int n = blockIdx.x, k = threadIdx.x;
  const float* W = nullptr; const float* bs = nullptr; int scol = 0, srcN = 0;
  if (n < 1024)       { int j = n >> 2, g = n & 3;            W = Wihf; scol = g*256 + j; srcN = 1024; bs = bf_;  }
  else if (n < 2048)  { int m = n-1024, j = m >> 2, g = m & 3; W = Wihb; scol = g*256 + j; srcN = 1024; bs = bb_;  }
  else if (n < 3072)  { int m = n-2048, j = m >> 2, g = m & 3;
                        if (g < 3) { W = Wiou; scol = g*256 + j; srcN = 768; bs = biou; } }
  else                { int m = n-3072;                        W = Wtf;  scol = m;       srcN = 256;  bs = btf;  }
  float v = 0.f, bv = 0.f;
  if (W && k < E) v = W[(size_t)k * srcN + scol];
  if (W) bv = bs[scol];
  if (k < EP) Wt[(size_t)n * EP + k] = (bf16_t)v;
  if (k == 0) bias[n] = bv;
}

// W_hh (fwd+bwd) -> [dir][N=1024 interleaved][K=256] bf16
__global__ void k_prep_whh(const float* Whhf, const float* Whhb, bf16_t* Wt) {
  int blk = blockIdx.x, dir = blk >> 10, n = blk & 1023, k = threadIdx.x;
  int scol = (n & 3) * 256 + (n >> 2);
  const float* W = dir ? Whhb : Whhf;
  Wt[((size_t)dir * 1024 + n) * 256 + k] = (bf16_t)W[(size_t)k * 1024 + scol];
}

// U_iou -> [1024 padded interleaved][256]; U_tf -> [256][256] transposed
__global__ void k_prep_u(const float* Uiou, const float* Utf, bf16_t* UiouT, bf16_t* UtfT) {
  int blk = blockIdx.x, k = threadIdx.x;
  if (blk < 1024) {
    int j = blk >> 2, g = blk & 3;
    float v = (g < 3) ? Uiou[(size_t)k * 768 + g * 256 + j] : 0.f;
    UiouT[(size_t)blk * 256 + k] = (bf16_t)v;
  } else {
    int n = blk - 1024;
    UtfT[(size_t)n * 256 + k] = (bf16_t)Utf[(size_t)k * 256 + n];
  }
}

// embedding gather, row r = t*B+b, padded to EP, bf16
__global__ void k_embed(const float* embed, const int* x, bf16_t* emb) {
  int r = blockIdx.x, e = threadIdx.x;
  if (e >= EP) return;
  int tok = x[r];
  float v = (e < E) ? embed[(size_t)tok * E + e] : 0.f;
  emb[(size_t)r * EP + e] = (bf16_t)v;
}

__global__ void k_zero(float* p, size_t n) {
  size_t i = (size_t)blockIdx.x * blockDim.x + threadIdx.x;
  if (i < n) p[i] = 0.f;
}

// ---------------- big projection GEMM: C[32768,3328] = emb @ Wbig + bias (bf16 WMMA, f32 acc) ----------------
__global__ void __launch_bounds__(256) k_gemm(const bf16_t* A, const bf16_t* Bt, const float* bias, bf16_t* C) {
  int tid = threadIdx.x, w = tid >> 5, lane = tid & 31, hi = lane >> 4, cc = lane & 15;
  int mb = blockIdx.x * 128 + (w >> 1) * 32;
  int nb = blockIdx.y * 64  + (w & 1) * 32;
  v8f acc[2][2];
#pragma unroll
  for (int i = 0; i < 2; i++)
#pragma unroll
    for (int j = 0; j < 2; j++) {
      float bv = bias[nb + j * 16 + cc];
#pragma unroll
      for (int r = 0; r < 8; r++) acc[i][j][r] = bv;
    }
#pragma unroll
  for (int kb = 0; kb < EP / 32; kb++) {
    v16bf a0 = loadA(A + (size_t)(mb + cc) * EP,      kb, hi);
    v16bf a1 = loadA(A + (size_t)(mb + 16 + cc) * EP, kb, hi);
    v16bf b0 = loadB(Bt + (size_t)(nb + cc) * EP,      kb, hi);
    v16bf b1 = loadB(Bt + (size_t)(nb + 16 + cc) * EP, kb, hi);
    acc[0][0] = wmma_bf16(a0, b0, acc[0][0]);
    acc[0][1] = wmma_bf16(a0, b1, acc[0][1]);
    acc[1][0] = wmma_bf16(a1, b0, acc[1][0]);
    acc[1][1] = wmma_bf16(a1, b1, acc[1][1]);
  }
#pragma unroll
  for (int i = 0; i < 2; i++)
#pragma unroll
    for (int j = 0; j < 2; j++)
#pragma unroll
      for (int r = 0; r < 8; r++) {
        int row = mb + i * 16 + hi * 8 + r, col = nb + j * 16 + cc;
        C[(size_t)row * NC4 + col] = (bf16_t)acc[i][j][r];
      }
}

// ---------------- sequential BiLSTM scans: one WGP per direction, h in LDS (double-buffered) ----------------
// A-fragments of h are hoisted out of the N-tile loop (depend only on kb), so each serial step is
// 8 LDS fragment loads + per-tile {8 seed loads, 8 (loadB+wmma), gate math with 4 wave32 shuffles}.
__global__ void __launch_bounds__(256) k_lstm(const bf16_t* Cbf, const bf16_t* Whh_all,
                                              const int* lengths, float* outF, float* outB, float* cws) {
  extern __shared__ __align__(16) bf16_t hdyn[];   // [2][64][256] bf16
  __shared__ int lenS[64];
  int dir = blockIdx.x;
  const bf16_t* Whh = Whh_all + (size_t)dir * 1024 * 256;
  float* c_dir = cws + (size_t)dir * 64 * 256;
  float* outp  = dir ? outB : outF;
  int colbase  = dir ? 1024 : 0;
  int tid = threadIdx.x, w = tid >> 5, lane = tid & 31, hi = lane >> 4, cc = lane & 15;
  for (int i = tid; i < 64 * 256; i += 256) { hdyn[i] = (bf16_t)0.f; c_dir[i] = 0.f; }
  if (tid < 64) lenS[tid] = lengths[tid];
  __syncthreads();
  int mt = w >> 1, ntbase = (w & 1) * 32;
  for (int t = 0; t < L; t++) {
    int cur = t & 1;
    bf16_t* hc = hdyn + cur * 64 * 256;
    bf16_t* hn = hdyn + (cur ^ 1) * 64 * 256;
    // per-step, per-lane row constants (xrow doubles as both xg-source row and output row)
    int   len8[8]; bool  m8[8];
    size_t srcoff[8], outoff[8];
#pragma unroll
    for (int r = 0; r < 8; r++) {
      int br = mt * 16 + hi * 8 + r;
      int len = lenS[br];
      int xrow = dir ? ((t < len) ? (len - 1 - t) : t) : t;
      len8[r]  = len;
      m8[r]    = (t < len);
      srcoff[r] = ((size_t)xrow * 64 + br) * NC4 + colbase;
      outoff[r] = ((size_t)xrow * 64 + br) * 256;
    }
    // hoisted A fragments of h (shared by all 32 N-tiles this wave owns)
    v16bf afr[8];
#pragma unroll
    for (int kb = 0; kb < 8; kb++) afr[kb] = loadA(hc + (size_t)(mt * 16 + cc) * 256, kb, hi);
    for (int nti = 0; nti < 32; nti++) {
      int nt = ntbase + nti, n = nt * 16 + cc;
      v8f acc;
#pragma unroll
      for (int r = 0; r < 8; r++) acc[r] = (float)Cbf[srcoff[r] + n];
#pragma unroll
      for (int kb = 0; kb < 8; kb++)
        acc = wmma_bf16(afr[kb], loadB(Whh + (size_t)n * 256, kb, hi), acc);
      int g = n & 3, jc = n >> 2, bl = lane & ~3;
#pragma unroll
      for (int r = 0; r < 8; r++) {
        int br = mt * 16 + hi * 8 + r;
        float val = acc[r];
        float act = (g == 2) ? tanhfast(val) : sigf(val);
        float gi = __shfl(act, bl + 0, 32);
        float gf = __shfl(act, bl + 1, 32);
        float gu = __shfl(act, bl + 2, 32);
        float go = __shfl(act, bl + 3, 32);
        if (g == 0) {
          bool m = m8[r];
          float cold = c_dir[br * 256 + jc];
          float cn = gf * cold + gi * gu;
          float hv = go * tanhfast(cn);
          c_dir[br * 256 + jc] = m ? cn : cold;
          float hold = (float)hc[br * 256 + jc];
          hn[br * 256 + jc] = (bf16_t)(m ? hv : hold);
          outp[outoff[r] + jc] = m ? hv : 0.f;
        }
      }
      (void)len8;
    }
    __syncthreads();
  }
}

// ---------------- tree-LSTM: 4 blocks x 16 batches, 512 serial steps, M=16 WMMA tiles ----------------
__global__ void __launch_bounds__(256) k_tree(const bf16_t* Cbf, const bf16_t* UiouT, const bf16_t* UtfT,
                                              const int* parents, const int* lengths,
                                              float* hsum, float* csum, float* treeh) {
  __shared__ bf16_t hstage[16 * 256];
  __shared__ bf16_t hnew[16 * 256];
  __shared__ float  hnewf[16 * 256];
  __shared__ float  cstep[16 * 256];
  __shared__ int pS[16], lenS[16];
  int b0 = blockIdx.x * 16;
  int tid = threadIdx.x, w = tid >> 5, lane = tid & 31, hi = lane >> 4, cc = lane & 15;
  if (tid < 16) lenS[tid] = lengths[b0 + tid];
  __syncthreads();
  for (int jj = 0; jj < L; jj++) {
    int j = L - 1 - jj;
    if (tid < 16) pS[tid] = parents[j * 64 + b0 + tid];
    for (int i = tid; i < 16 * 256; i += 256) {
      int bb = i >> 8, col = i & 255;
      hstage[i] = (bf16_t)hsum[((size_t)(b0 + bb) * L + j) * 256 + col];
    }
    __syncthreads();
    // iou = iou_x[:,j,:] + h_sum[:,j,:] @ U_iou'   (gate-interleaved, padded to 1024)
    {
      size_t srcoff[8], csoff[8];
#pragma unroll
      for (int r = 0; r < 8; r++) {
        int bb = hi * 8 + r;
        srcoff[r] = ((size_t)j * 64 + b0 + bb) * NC4 + 2048;
        csoff[r]  = ((size_t)(b0 + bb) * L + j) * 256;
      }
      v16bf afr[8];
#pragma unroll
      for (int kb = 0; kb < 8; kb++) afr[kb] = loadA(hstage + cc * 256, kb, hi);
      for (int q = 0; q < 8; q++) {
        int nt = w * 8 + q, n = nt * 16 + cc, g = n & 3, jh = n >> 2, bl = lane & ~3;
        v8f acc;
#pragma unroll
        for (int r = 0; r < 8; r++) acc[r] = (float)Cbf[srcoff[r] + n];
#pragma unroll
        for (int kb = 0; kb < 8; kb++)
          acc = wmma_bf16(afr[kb], loadB(UiouT + (size_t)n * 256, kb, hi), acc);
#pragma unroll
        for (int r = 0; r < 8; r++) {
          int bb = hi * 8 + r;
          float val = acc[r];
          float act = (g == 2) ? tanhfast(val) : sigf(val);
          float gi = __shfl(act, bl + 0, 32);
          float go = __shfl(act, bl + 1, 32);
          float gu = __shfl(act, bl + 2, 32);
          if (g == 0) {
            float cs = csum[csoff[r] + jh];
            float cv = gi * gu + cs;
            float hv = go * tanhfast(cv);
            cstep[bb * 256 + jh] = cv;
            hnewf[bb * 256 + jh] = hv;
            hnew[bb * 256 + jh]  = (bf16_t)hv;
          }
        }
      }
    }
    __syncthreads();
    // f = sigmoid(f_x[:,parent,:] + h @ U_tf); scatter-add into parent rows
    {
      v16bf afr[8];
#pragma unroll
      for (int kb = 0; kb < 8; kb++) afr[kb] = loadA(hnew + cc * 256, kb, hi);
      for (int q = 0; q < 2; q++) {
        int nt2 = w * 2 + q, colh = nt2 * 16 + cc;
        v8f acc;
#pragma unroll
        for (int r = 0; r < 8; r++) {
          int bb = hi * 8 + r, p = pS[bb];
          acc[r] = (float)Cbf[((size_t)p * 64 + b0 + bb) * NC4 + 3072 + colh];
        }
#pragma unroll
        for (int kb = 0; kb < 8; kb++)
          acc = wmma_bf16(afr[kb], loadB(UtfT + (size_t)colh * 256, kb, hi), acc);
#pragma unroll
        for (int r = 0; r < 8; r++) {
          int bb = hi * 8 + r, p = pS[bb];
          float fs = sigf(acc[r]);
          float fc = fs * cstep[bb * 256 + colh];
          bool valid = (j > 0) && (j < lenS[bb]);
          size_t go = ((size_t)(b0 + bb) * L + p) * 256 + colh;
          if (valid) { hsum[go] += hnewf[bb * 256 + colh]; csum[go] += fc; }
          if (j == 0) treeh[(size_t)(b0 + bb) * 256 + colh] = hnewf[bb * 256 + colh];
        }
      }
    }
    __threadfence();
    __syncthreads();
  }
}

// ---------------- attention + classifier: one block per batch ----------------
__global__ void __launch_bounds__(256) k_attn(const float* treeh, const float* Wlin, const float* blin,
                                              const float* Wout, const float* bout,
                                              const float* outF, const float* outB, const int* lengths,
                                              float* out) {
  __shared__ float thv[256], th[256], al[512], red[256], ctx[256];
  int b = blockIdx.x, tid = threadIdx.x;
  thv[tid] = treeh[(size_t)b * 256 + tid];
  __syncthreads();
  float s = blin[tid];
  for (int k = 0; k < 256; k++) s += thv[k] * Wlin[(size_t)k * 256 + tid];
  th[tid] = s;
  __syncthreads();
  int len = lengths[b];
  for (int t = tid; t < 512; t += 256) {
    float a = -1e30f;
    if (t < len) {
      const float* of = outF + ((size_t)t * 64 + b) * 256;
      const float* ob = outB + ((size_t)t * 64 + b) * 256;
      a = 0.f;
      for (int h = 0; h < 256; h++) a += th[h] * (of[h] + ob[h]);
    }
    al[t] = a;
  }
  __syncthreads();
  red[tid] = fmaxf(al[tid], al[tid + 256]);
  __syncthreads();
  for (int st = 128; st > 0; st >>= 1) { if (tid < st) red[tid] = fmaxf(red[tid], red[tid + st]); __syncthreads(); }
  float mx = red[0];
  __syncthreads();
  float e0 = __expf(al[tid] - mx), e1 = __expf(al[tid + 256] - mx);
  al[tid] = e0; al[tid + 256] = e1;
  red[tid] = e0 + e1;
  __syncthreads();
  for (int st = 128; st > 0; st >>= 1) { if (tid < st) red[tid] += red[tid + st]; __syncthreads(); }
  float inv = 1.f / red[0];
  __syncthreads();
  float cv = 0.f;
  for (int t = 0; t < 512; t++)
    cv += al[t] * (outF[((size_t)t * 64 + b) * 256 + tid] + outB[((size_t)t * 64 + b) * 256 + tid]);
  ctx[tid] = cv * inv;
  __syncthreads();
  for (int c = 0; c < 2; c++) {
    red[tid] = thv[tid] * Wout[(size_t)tid * 2 + c] + ctx[tid] * Wout[(size_t)(256 + tid) * 2 + c];
    __syncthreads();
    for (int st = 128; st > 0; st >>= 1) { if (tid < st) red[tid] += red[tid + st]; __syncthreads(); }
    if (tid == 0) out[b * 2 + c] = sigf(red[0] + bout[c]);
    __syncthreads();
  }
}

// ---------------- launch ----------------
extern "C" void kernel_launch(void* const* d_in, const int* in_sizes, int n_in,
                              void* d_out, int out_size, void* d_ws, size_t ws_size,
                              hipStream_t stream) {
  const float* embed   = (const float*)d_in[0];
  const float* Wihf    = (const float*)d_in[1];
  const float* Whhf    = (const float*)d_in[2];
  const float* bf_     = (const float*)d_in[3];
  const float* Wihb    = (const float*)d_in[4];
  const float* Whhb    = (const float*)d_in[5];
  const float* bb_     = (const float*)d_in[6];
  const float* Wiou    = (const float*)d_in[7];
  const float* Uiou    = (const float*)d_in[8];
  const float* biou    = (const float*)d_in[9];
  const float* Wtf     = (const float*)d_in[10];
  const float* Utf     = (const float*)d_in[11];
  const float* btf     = (const float*)d_in[12];
  const float* Wlin    = (const float*)d_in[13];
  const float* blin    = (const float*)d_in[14];
  const float* Wout    = (const float*)d_in[15];
  const float* bout    = (const float*)d_in[16];
  const int*   x       = (const int*)d_in[17];
  const int*   parents = (const int*)d_in[18];
  const int*   lengths = (const int*)d_in[19];

  char* ws = (char*)d_ws;
  bf16_t* emb   = (bf16_t*)(ws + OFF_EMB);
  bf16_t* Wbig  = (bf16_t*)(ws + OFF_WBIG);
  float*  bias  = (float*) (ws + OFF_BIAS);
  bf16_t* Cbf   = (bf16_t*)(ws + OFF_C);
  bf16_t* Whh   = (bf16_t*)(ws + OFF_WHH);
  bf16_t* UiouT = (bf16_t*)(ws + OFF_UIOU);
  bf16_t* UtfT  = (bf16_t*)(ws + OFF_UTF);
  float*  outF  = (float*) (ws + OFF_OUTF);
  float*  outB  = (float*) (ws + OFF_OUTB);
  float*  hsum  = (float*) (ws + OFF_HSUM);
  float*  csum  = (float*) (ws + OFF_CSUM);
  float*  cst   = (float*) (ws + OFF_CST);
  float*  treeh = (float*) (ws + OFF_TH);

  k_prep_wbig<<<NC4, 256, 0, stream>>>(Wihf, bf_, Wihb, bb_, Wiou, biou, Wtf, btf, Wbig, bias);
  k_prep_whh<<<2048, 256, 0, stream>>>(Whhf, Whhb, Whh);
  k_prep_u<<<1280, 256, 0, stream>>>(Uiou, Utf, UiouT, UtfT);
  k_embed<<<ROWS, 256, 0, stream>>>(embed, x, emb);
  k_gemm<<<dim3(ROWS / 128, NC4 / 64), 256, 0, stream>>>(emb, Wbig, bias, Cbf);
  k_zero<<<(unsigned)((2ull * B * L * H + 255) / 256), 256, 0, stream>>>(hsum, (size_t)2 * B * L * H);
  k_lstm<<<2, 256, 2 * 64 * 256 * sizeof(bf16_t), stream>>>(Cbf, Whh, lengths, outF, outB, cst);
  k_tree<<<4, 256, 0, stream>>>(Cbf, UiouT, UtfT, parents, lengths, hsum, csum, treeh);
  k_attn<<<64, 256, 0, stream>>>(treeh, Wlin, blin, Wout, bout, outF, outB, lengths, (float*)d_out);

  (void)in_sizes; (void)n_in; (void)out_size; (void)ws_size;
}
